// GraphLayer_58978490908834
// MI455X (gfx1250) — matrix operand order repaired and checked by reading, not compiled, not run on hardware
//
#include <hip/hip_runtime.h>
#include <hip/hip_bf16.h>

// CDNA5 / gfx1250 wave32 WMMA types
typedef __attribute__((ext_vector_type(16))) _Float16 v16h;
typedef __attribute__((ext_vector_type(8)))  _Float16 v8h;
typedef __attribute__((ext_vector_type(8)))  float    v8f;

#define DDIM   32      // node feature dim (== out dim)
#define DEDIM  16      // edge feature dim
#define DHDIM  64      // hidden dim
#define DINDIM 80      // 2*D + DE
#define KCH1   3       // ceil(80/32) K-chunks for GEMM1 (padded to 96)
#define NT1    4       // 64/16 N-tiles for GEMM1
#define KCH2   2       // 64/32 K-chunks for GEMM2
#define NT2    2       // 32/16 N-tiles for GEMM2
#define LDS_ROW 72     // 64 halves + 8 pad (144B rows, 16B aligned)

__global__ void zero_f32_kernel(float* __restrict__ p, int n) {
    int i = blockIdx.x * blockDim.x + threadIdx.x;
    if (i < n) p[i] = 0.0f;
}

// One wave processes 16 edges at a time:
//   A[16,80(->96)] (gathered, f32->f16)  @  W1[96,64]  -> C1[16,64]  (12 wmma)
//   relu(C1+b1) -> LDS (f16) -> A2 fragments
//   A2[16,64] @ W2[64,32] -> C2[16,32]  (4 wmma), + b2, atomic scatter to out[dst]
__launch_bounds__(256)
__global__ void edge_mlp_scatter_kernel(
    const float* __restrict__ hin,     // [N, 32]
    const float* __restrict__ efeat,   // [E, 16]
    const int*   __restrict__ srcIdx,  // [E]
    const int*   __restrict__ dstIdx,  // [E]
    const float* __restrict__ W1,      // [80, 64] row-major
    const float* __restrict__ b1,      // [64]
    const float* __restrict__ W2,      // [64, 32] row-major
    const float* __restrict__ b2,      // [32]
    float*       __restrict__ out,     // [N, 32], pre-zeroed, atomic accumulate
    int E)
{
    __shared__ __align__(16) _Float16 ldsA2[8][16 * LDS_ROW]; // per-wave f16 scratch
    __shared__ int ldsDst[8][16];                             // per-wave dst indices

    const int  lane = threadIdx.x & 31;
    const int  wave = threadIdx.x >> 5;
    const bool hi   = lane >= 16;
    const int  col  = lane & 15;

    // ---------------- Load weight B-fragments once per wave ----------------
    // B layout (32x16 f16, 8 VGPRs): within a lane, half t ->
    // K = t + (t>=8 ? 8 : 0) + (hi ? 8 : 0); N = lane%16.
    v16h bw1[KCH1][NT1];
#pragma unroll
    for (int kc = 0; kc < KCH1; ++kc) {
#pragma unroll
        for (int nt = 0; nt < NT1; ++nt) {
#pragma unroll
            for (int t = 0; t < 16; ++t) {
                int k = kc * 32 + t + ((t >= 8) ? 8 : 0) + (hi ? 8 : 0);
                int n = nt * 16 + col;
                float w = (k < DINDIM) ? W1[k * DHDIM + n] : 0.0f;
                bw1[kc][nt][t] = (_Float16)w;
            }
        }
    }
    v16h bw2[KCH2][NT2];
#pragma unroll
    for (int kc = 0; kc < KCH2; ++kc) {
#pragma unroll
        for (int nt = 0; nt < NT2; ++nt) {
#pragma unroll
            for (int t = 0; t < 16; ++t) {
                int k = kc * 32 + t + ((t >= 8) ? 8 : 0) + (hi ? 8 : 0);
                int n = nt * 16 + col;
                bw2[kc][nt][t] = (_Float16)W2[k * DDIM + n];
            }
        }
    }
    float b1c[NT1];
#pragma unroll
    for (int nt = 0; nt < NT1; ++nt) b1c[nt] = b1[nt * 16 + col];
    float b2c[NT2];
#pragma unroll
    for (int nt = 0; nt < NT2; ++nt) b2c[nt] = b2[nt * 16 + col];

    const int nTiles = (E + 15) >> 4;
    const int waveId = blockIdx.x * (blockDim.x >> 5) + wave;
    const int nWaves = gridDim.x * (blockDim.x >> 5);

    for (int tile = waveId; tile < nTiles; tile += nWaves) {
        // Each lane owns edge row m = lane%16 for the A fragment.
        long e = (long)tile * 16 + col;
        if (e >= E) e = E - 1;                    // clamp for tail tile (gather only)
        const int sIdx = srcIdx[e];
        const int dIdx = dstIdx[e];
        if (!hi) ldsDst[wave][col] = dIdx;

        // Prefetch next tile's indices (global_prefetch_b8)
        int nextTile = tile + nWaves;
        if (nextTile < nTiles) {
            __builtin_prefetch(&srcIdx[(long)nextTile * 16 + col], 0, 0);
            __builtin_prefetch(&dstIdx[(long)nextTile * 16 + col], 0, 0);
        }

        // ---------------- GEMM1: gather A chunks, 12 WMMAs ----------------
        v8f acc[NT1];
#pragma unroll
        for (int nt = 0; nt < NT1; ++nt)
#pragma unroll
            for (int r = 0; r < 8; ++r) acc[nt][r] = 0.0f;

#pragma unroll
        for (int kc = 0; kc < KCH1; ++kc) {
            // A layout (16x32 f16): half t -> K = kc*32 + t + (t>=8?8:0) + (hi?8:0)
            v16h a;
#pragma unroll
            for (int t = 0; t < 16; ++t) {
                int kl = t + ((t >= 8) ? 8 : 0) + (hi ? 8 : 0);
                int k  = kc * 32 + kl;
                float v;
                if (k < DDIM)            v = hin[(long)dIdx * DDIM + k];
                else if (k < 2 * DDIM)   v = hin[(long)sIdx * DDIM + (k - DDIM)];
                else if (k < DINDIM)     v = efeat[e * DEDIM + (k - 2 * DDIM)];
                else                     v = 0.0f;
                a[t] = (_Float16)v;
            }
#pragma unroll
            for (int nt = 0; nt < NT1; ++nt) {
                acc[nt] = __builtin_amdgcn_wmma_f32_16x16x32_f16(
                    false, a, false, bw1[kc][nt], (short)0, acc[nt], false, false);
            }
        }

        // -------- bias + ReLU, C-layout -> f16 LDS tile [m][64] ----------
        // C/D layout: lane holds N = col, rows M = r + (hi?8:0) in VGPR r.
#pragma unroll
        for (int nt = 0; nt < NT1; ++nt) {
#pragma unroll
            for (int r = 0; r < 8; ++r) {
                float v = acc[nt][r] + b1c[nt];
                v = v > 0.0f ? v : 0.0f;
                int m = r + (hi ? 8 : 0);
                ldsA2[wave][m * LDS_ROW + nt * 16 + col] = (_Float16)v;
            }
        }
        // Same-wave DS ops complete in order; compiler sees the dependency.

        // -------- GEMM2: A2 fragments from LDS (aligned b128 reads) -------
        v16h a2[KCH2];
#pragma unroll
        for (int kc = 0; kc < KCH2; ++kc) {
            const _Float16* base = &ldsA2[wave][col * LDS_ROW + kc * 32 + (hi ? 8 : 0)];
            v8h lo = *(const v8h*)(base);        // K halves t=0..7
            v8h ho = *(const v8h*)(base + 16);   // K halves t=8..15
            a2[kc] = __builtin_shufflevector(lo, ho, 0, 1, 2, 3, 4, 5, 6, 7,
                                             8, 9, 10, 11, 12, 13, 14, 15);
        }
        v8f acc2[NT2];
#pragma unroll
        for (int nt = 0; nt < NT2; ++nt)
#pragma unroll
            for (int r = 0; r < 8; ++r) acc2[nt][r] = 0.0f;
#pragma unroll
        for (int kc = 0; kc < KCH2; ++kc) {
#pragma unroll
            for (int nt = 0; nt < NT2; ++nt) {
                acc2[nt] = __builtin_amdgcn_wmma_f32_16x16x32_f16(
                    false, a2[kc], false, bw2[kc][nt], (short)0, acc2[nt], false, false);
            }
        }

        // ---------------- bias + atomic scatter-add ----------------
#pragma unroll
        for (int nt = 0; nt < NT2; ++nt) {
#pragma unroll
            for (int r = 0; r < 8; ++r) {
                int  m  = r + (hi ? 8 : 0);
                long eg = (long)tile * 16 + m;
                if (eg < E) {
                    int d = ldsDst[wave][m];
                    atomicAdd(&out[(long)d * DDIM + nt * 16 + col], acc2[nt][r] + b2c[nt]);
                }
            }
        }
    }
}

extern "C" void kernel_launch(void* const* d_in, const int* in_sizes, int n_in,
                              void* d_out, int out_size, void* d_ws, size_t ws_size,
                              hipStream_t stream) {
    const float* h     = (const float*)d_in[0];   // [N, 32]
    const float* edges = (const float*)d_in[1];   // [2, E, 16]
    const int*   eidx  = (const int*)d_in[2];     // [2, 2, E]
    const float* W1    = (const float*)d_in[3];   // [80, 64]
    const float* b1    = (const float*)d_in[4];   // [64]
    const float* W2    = (const float*)d_in[5];   // [64, 32]
    const float* b2    = (const float*)d_in[6];   // [32]

    float* out = (float*)d_out;                   // [N, 32]
    float* h1  = (float*)d_ws;                    // [N, 32] intermediate

    const int E = in_sizes[2] / 4;                // edge_index: 2*2*E ints
    const int Nn = in_sizes[0] / DDIM;            // nodes
    const int nElems = Nn * DDIM;

    const int zblk = (nElems + 255) / 256;
    const int nTiles = (E + 15) / 16;
    int blocks = (nTiles + 7) / 8;                // 8 waves per 256-thread block
    if (blocks > 4096) blocks = 4096;

    // Pass 1: h -> h1 using edge set 0
    zero_f32_kernel<<<zblk, 256, 0, stream>>>(h1, nElems);
    edge_mlp_scatter_kernel<<<blocks, 256, 0, stream>>>(
        h, edges, eidx, eidx + E, W1, b1, W2, b2, h1, E);

    // Pass 2: h1 -> out using edge set 1
    zero_f32_kernel<<<zblk, 256, 0, stream>>>(out, nElems);
    edge_mlp_scatter_kernel<<<blocks, 256, 0, stream>>>(
        h1, edges + (long)E * DEDIM, eidx + 2L * E, eidx + 3L * E,
        W1, b1, W2, b2, out, E);
}
